// HierachicalEncoder_6425271075473
// MI455X (gfx1250) — matrix operand order repaired and checked by reading, not compiled, not run on hardware
//
#include <hip/hip_runtime.h>
#include <hip/hip_bf16.h>
#include <stdint.h>

// ---------------------------------------------------------------------------
// Types for CDNA5 WMMA
// ---------------------------------------------------------------------------
typedef __bf16 bf16_t;
typedef __attribute__((ext_vector_type(16))) __bf16 v16bf;
typedef __attribute__((ext_vector_type(8)))  float  v8f;

#define NUM_ITEM 20000
#define DIM_D 64

union FragBF {
    v16bf  v;
    bf16_t h[16];
    uint4  q[2];
};

__device__ __forceinline__ bf16_t f2bf(float f) {
    union { unsigned short u; bf16_t b; } cv;
    unsigned x = __float_as_uint(f);
    cv.u = (unsigned short)((x + 0x7FFFu + ((x >> 16) & 1u)) >> 16);  // RNE
    return cv.b;
}

// ---------------------------------------------------------------------------
// fp32 -> bf16 elementwise convert
// ---------------------------------------------------------------------------
__global__ __launch_bounds__(256) void cvt_f32_bf16_kernel(
    const float* __restrict__ src, bf16_t* __restrict__ dst, int n)
{
    int i = blockIdx.x * 256 + threadIdx.x;
    if (i < n) dst[i] = f2bf(src[i]);
}

// ---------------------------------------------------------------------------
// Per-row L2 normalize (x / max(||x||, 1e-12)), fp32 -> bf16
// one block (256 threads) per row
// ---------------------------------------------------------------------------
__global__ __launch_bounds__(256) void l2norm_rows_kernel(
    const float* __restrict__ src, bf16_t* __restrict__ dst, int C)
{
    const int row = blockIdx.x;
    const float* s = src + (size_t)row * C;
    float partial = 0.f;
    for (int i = threadIdx.x; i < C; i += 256) { float v = s[i]; partial += v * v; }
#pragma unroll
    for (int m = 16; m >= 1; m >>= 1) partial += __shfl_xor(partial, m, 32);
    __shared__ float red[8];
    if ((threadIdx.x & 31) == 0) red[threadIdx.x >> 5] = partial;
    __syncthreads();
    float total = 0.f;
#pragma unroll
    for (int w = 0; w < 8; ++w) total += red[w];
    const float scale = 1.0f / fmaxf(sqrtf(total), 1e-12f);
    bf16_t* d = dst + (size_t)row * C;
    for (int i = threadIdx.x; i < C; i += 256) d[i] = f2bf(s[i] * scale);
}

// ---------------------------------------------------------------------------
// WMMA GEMM:  C[M,N] = act( A[M,K](bf16,row-major) * W[N,K](bf16)^T + bias[N] )
//
// Block: 256 threads = 8 waves stacked in M -> tile 256(M) x 64(N).
// Per wave per k-step: 2 A fragments (32 rows), 4 B fragments, 8 WMMAs.
// Weight tile 64x32 bf16 double-buffered in LDS (one barrier per k-step):
//   regs <- gmem(k+32);  compute from lds[cur];  lds[next] <- regs;  barrier.
// All 4 B fragments are loaded before any WMMA so the scheduler batches the
// ds_load_b128s under one s_wait_dscnt.
// Rows are clamped so EXEC stays all-1s at the WMMA (ISA requirement).
// ---------------------------------------------------------------------------
template <bool RELU, bool OUT_BF16>
__global__ __launch_bounds__(256) void gemm_bf16_wmma_kernel(
    const bf16_t* __restrict__ A,
    const bf16_t* __restrict__ W,
    const float*  __restrict__ bias,
    void* __restrict__ Cout,
    int M, int N, int K)
{
    __shared__ bf16_t wtile[2][64 * 32];       // 2 x 4 KB weight stages

    const int tid   = threadIdx.x;
    const int lane  = tid & 31;
    const int wave  = tid >> 5;
    const int mBase = blockIdx.x * 256 + wave * 32;
    const int nBase = blockIdx.y * 64;

    const int mrow = lane & 15;
    const int half = lane >> 4;                // 0: lanes 0-15, 1: lanes 16-31

    int arow0 = mBase + mrow;       if (arow0 >= M) arow0 = M - 1;
    int arow1 = mBase + 16 + mrow;  if (arow1 >= M) arow1 = M - 1;
    const bf16_t* Arow0 = A + (size_t)arow0 * K;
    const bf16_t* Arow1 = A + (size_t)arow1 * K;

    // weight staging: thread i moves one 16B chunk (64 rows x 4 chunks)
    const int srow = tid >> 2;
    const int sseg = tid & 3;
    const bf16_t* Wsrc  = W + (size_t)(nBase + srow) * K + sseg * 8;
    bf16_t* wdst[2] = { &wtile[0][srow * 32 + sseg * 8],
                        &wtile[1][srow * 32 + sseg * 8] };

    v8f acc[2][4];
#pragma unroll
    for (int mt = 0; mt < 2; ++mt)
#pragma unroll
        for (int nt = 0; nt < 4; ++nt)
            acc[mt][nt] = (v8f){0.f,0.f,0.f,0.f,0.f,0.f,0.f,0.f};

    // prologue: stage k=0 tile into buffer 0
    *reinterpret_cast<uint4*>(wdst[0]) = *reinterpret_cast<const uint4*>(Wsrc);
    __syncthreads();

    for (int k = 0; k < K; k += 32) {
        const int  cur     = (k >> 5) & 1;
        const bool hasNext = (k + 32) < K;
        const bf16_t* wbuf = wtile[cur];

        // prefetch next weight tile into registers (hides gmem latency)
        uint4 stg;
        if (hasNext) {
            stg = *reinterpret_cast<const uint4*>(Wsrc + k + 32);
            __builtin_prefetch(Arow0 + k + 64, 0, 3);   // global_prefetch_b8
            __builtin_prefetch(Arow1 + k + 64, 0, 3);
        }

        // ---- A fragments (16x32 bf16 each; lane half selects K sub-block) --
        FragBF fa0, fa1;
        fa0.q[0] = *reinterpret_cast<const uint4*>(Arow0 + k + half * 8);
        fa0.q[1] = *reinterpret_cast<const uint4*>(Arow0 + k + 16 + half * 8);
        fa1.q[0] = *reinterpret_cast<const uint4*>(Arow1 + k + half * 8);
        fa1.q[1] = *reinterpret_cast<const uint4*>(Arow1 + k + 16 + half * 8);

        // ---- all 4 B fragments from LDS, then 8 WMMAs ----
        FragBF fb[4];
#pragma unroll
        for (int nt = 0; nt < 4; ++nt) {
            const bf16_t* src = &wbuf[(nt * 16 + mrow) * 32 + half * 8];
            fb[nt].q[0] = *reinterpret_cast<const uint4*>(src);
            fb[nt].q[1] = *reinterpret_cast<const uint4*>(src + 16);
        }
#pragma unroll
        for (int nt = 0; nt < 4; ++nt) {
            acc[0][nt] = __builtin_amdgcn_wmma_f32_16x16x32_bf16(
                false, fa0.v, false, fb[nt].v, (short)0, acc[0][nt], false, false);
            acc[1][nt] = __builtin_amdgcn_wmma_f32_16x16x32_bf16(
                false, fa1.v, false, fb[nt].v, (short)0, acc[1][nt], false, false);
        }

        // ---- write next tile to the other buffer, one barrier per step ----
        if (hasNext) *reinterpret_cast<uint4*>(wdst[cur ^ 1]) = stg;
        __syncthreads();
    }

    // ---- epilogue: bias (+ReLU), guarded stores ----
    const int nLocal = lane & 15;
#pragma unroll
    for (int mt = 0; mt < 2; ++mt) {
#pragma unroll
        for (int nt = 0; nt < 4; ++nt) {
            const int n = nBase + nt * 16 + nLocal;
            const float b = bias[n];
#pragma unroll
            for (int r = 0; r < 8; ++r) {
                const int m = mBase + mt * 16 + half * 8 + r;
                if (m < M) {
                    float val = acc[mt][nt][r] + b;
                    if (RELU) val = fmaxf(val, 0.f);
                    if (OUT_BF16)
                        ((bf16_t*)Cout)[(size_t)m * N + n] = f2bf(val);
                    else
                        ((float*)Cout)[(size_t)m * N + n] = val;
                }
            }
        }
    }
}

// ---------------------------------------------------------------------------
// mm_full = l2(cfeat) + l2(tfeat); cf_full = cold ? mm_full : cf_lin
// one wave per item row (D=64 -> 2 elems/lane)
// ---------------------------------------------------------------------------
__global__ __launch_bounds__(256) void mm_combine_kernel(
    const float* __restrict__ cfeat, const float* __restrict__ tfeat,
    const float* __restrict__ cf_lin, const int* __restrict__ cold,
    float* __restrict__ mm_full, float* __restrict__ cf_full, int Nrows)
{
    const int row = blockIdx.x * 8 + (threadIdx.x >> 5);
    if (row >= Nrows) return;
    const int l = threadIdx.x & 31;
    const float* c = cfeat  + (size_t)row * 64;
    const float* t = tfeat  + (size_t)row * 64;
    const float* g = cf_lin + (size_t)row * 64;

    float c0 = c[l], c1 = c[l + 32], t0 = t[l], t1 = t[l + 32];
    float sc = c0 * c0 + c1 * c1;
    float st = t0 * t0 + t1 * t1;
#pragma unroll
    for (int m = 16; m >= 1; m >>= 1) { sc += __shfl_xor(sc, m, 32); st += __shfl_xor(st, m, 32); }
    const float rc = 1.f / fmaxf(sqrtf(sc), 1e-12f);
    const float rt = 1.f / fmaxf(sqrtf(st), 1e-12f);
    const float m0 = c0 * rc + t0 * rt;
    const float m1 = c1 * rc + t1 * rt;
    mm_full[(size_t)row * 64 + l]      = m0;
    mm_full[(size_t)row * 64 + l + 32] = m1;
    const bool isCold = cold[row] > 0;
    cf_full[(size_t)row * 64 + l]      = isCold ? m0 : g[l];
    cf_full[(size_t)row * 64 + l + 32] = isCold ? m1 : g[l + 32];
}

// ---------------------------------------------------------------------------
// Fused gather + L2 + LayerNorm + qkv + 3x3 softmax-attention + mean.
// One wave per token (lane owns channels l and l+32).
// w_q/w_k/w_v staged in LDS with stride 65 (pad) to avoid bank conflicts.
// ---------------------------------------------------------------------------
__global__ __launch_bounds__(256) void fused_attn_kernel(
    const int*   __restrict__ seq,
    const float* __restrict__ mm_full,
    const float* __restrict__ item_emb,
    const float* __restrict__ cf_full,
    const float* __restrict__ wq,
    const float* __restrict__ wk,
    const float* __restrict__ wv,
    float* __restrict__ out, int numTok)
{
    __shared__ float wL[3][64 * 65];       // padded: bank = (d + j) % 64
    __shared__ float xn[8][3][64];

    for (int i = threadIdx.x; i < 64 * 64; i += 256) {
        const int r = i >> 6, c = i & 63;
        wL[0][r * 65 + c] = wq[i];
        wL[1][r * 65 + c] = wk[i];
        wL[2][r * 65 + c] = wv[i];
    }
    __syncthreads();

    const int wave = threadIdx.x >> 5;
    const int l    = threadIdx.x & 31;
    const int t    = blockIdx.x * 8 + wave;
    if (t >= numTok) return;

    int s = seq[t];
    if (s == NUM_ITEM) s = 0;              // padding token -> index 0

    // gather + l2 + LayerNorm for the 3 modality rows
#pragma unroll
    for (int n = 0; n < 3; ++n) {
        const float* src = (n == 0) ? (mm_full + (size_t)s * 64)
                         : (n == 1) ? (item_emb + (size_t)s * 64)
                                    : (cf_full + (size_t)s * 64);
        float x0 = src[l], x1 = src[l + 32];
        float ss = x0 * x0 + x1 * x1;
#pragma unroll
        for (int m = 16; m >= 1; m >>= 1) ss += __shfl_xor(ss, m, 32);
        const float sc = 1.f / fmaxf(sqrtf(ss), 1e-12f);
        x0 *= sc; x1 *= sc;

        float sum = x0 + x1;
#pragma unroll
        for (int m = 16; m >= 1; m >>= 1) sum += __shfl_xor(sum, m, 32);
        const float mu = sum * (1.f / 64.f);
        const float d0 = x0 - mu, d1 = x1 - mu;
        float var = d0 * d0 + d1 * d1;
#pragma unroll
        for (int m = 16; m >= 1; m >>= 1) var += __shfl_xor(var, m, 32);
        const float inv = rsqrtf(var * (1.f / 64.f) + 1e-5f);
        xn[wave][n][l]      = d0 * inv;
        xn[wave][n][l + 32] = d1 * inv;
    }

    // q,k,v projections (lane's two channels), q pre-scaled by D^-0.5
    float q[3][2], kk[3][2], vv[3][2];
#pragma unroll
    for (int n = 0; n < 3; ++n) {
#pragma unroll
        for (int h = 0; h < 2; ++h) {
            const int d = l + h * 32;
            float aq = 0.f, ak = 0.f, av = 0.f;
#pragma unroll 8
            for (int j = 0; j < 64; ++j) {
                const float x = xn[wave][n][j];          // LDS broadcast
                aq += x * wL[0][d * 65 + j];
                ak += x * wL[1][d * 65 + j];
                av += x * wL[2][d * 65 + j];
            }
            q[n][h]  = aq * 0.125f;                      // 64^-0.5
            kk[n][h] = ak;
            vv[n][h] = av;
        }
    }

    // 3x3 scores via cross-lane reduction
    float p[3][3];
#pragma unroll
    for (int n = 0; n < 3; ++n)
#pragma unroll
        for (int m = 0; m < 3; ++m) {
            float sco = q[n][0] * kk[m][0] + q[n][1] * kk[m][1];
#pragma unroll
            for (int mk = 16; mk >= 1; mk >>= 1) sco += __shfl_xor(sco, mk, 32);
            p[n][m] = sco;
        }

    // softmax over m, y = attn @ v, mean over n
    float y0 = 0.f, y1 = 0.f;
#pragma unroll
    for (int n = 0; n < 3; ++n) {
        const float mx = fmaxf(p[n][0], fmaxf(p[n][1], p[n][2]));
        const float e0 = __expf(p[n][0] - mx);
        const float e1 = __expf(p[n][1] - mx);
        const float e2 = __expf(p[n][2] - mx);
        const float inv = 1.f / (e0 + e1 + e2);
        y0 += (e0 * vv[0][0] + e1 * vv[1][0] + e2 * vv[2][0]) * inv;
        y1 += (e0 * vv[0][1] + e1 * vv[1][1] + e2 * vv[2][1]) * inv;
    }
    out[(size_t)t * 64 + l]      = y0 * (1.f / 3.f);
    out[(size_t)t * 64 + l + 32] = y1 * (1.f / 3.f);
}

// ---------------------------------------------------------------------------
// Host orchestration
// ---------------------------------------------------------------------------
extern "C" void kernel_launch(void* const* d_in, const int* in_sizes, int n_in,
                              void* d_out, int out_size, void* d_ws, size_t ws_size,
                              hipStream_t stream)
{
    (void)in_sizes; (void)n_in; (void)out_size; (void)ws_size;

    const int*   seq      = (const int*)  d_in[0];
    const int*   cold     = (const int*)  d_in[1];
    const float* content  = (const float*)d_in[2];
    const float* text     = (const float*)d_in[3];
    const float* cf_feat  = (const float*)d_in[4];
    const float* item_emb = (const float*)d_in[5];
    const float* c_w1 = (const float*)d_in[6];  const float* c_b1 = (const float*)d_in[7];
    const float* c_w2 = (const float*)d_in[8];  const float* c_b2 = (const float*)d_in[9];
    const float* c_w3 = (const float*)d_in[10]; const float* c_b3 = (const float*)d_in[11];
    const float* t_w1 = (const float*)d_in[12]; const float* t_b1 = (const float*)d_in[13];
    const float* t_w2 = (const float*)d_in[14]; const float* t_b2 = (const float*)d_in[15];
    const float* t_w3 = (const float*)d_in[16]; const float* t_b3 = (const float*)d_in[17];
    const float* cf_w = (const float*)d_in[18]; const float* cf_b = (const float*)d_in[19];
    const float* w_q  = (const float*)d_in[20];
    const float* w_k  = (const float*)d_in[21];
    const float* w_v  = (const float*)d_in[22];
    float* out = (float*)d_out;

    const int N = NUM_ITEM;          // 20000 items
    const int TOK = 512 * 100;       // 51200 tokens

    // ---- workspace bump allocator (256B aligned) ----
    char* ws = (char*)d_ws;
    size_t off = 0;
    auto alloc = [&](size_t bytes) -> char* {
        char* p = ws + off;
        off += (bytes + 255) & ~(size_t)255;
        return p;
    };
    bf16_t* wb_cw1 = (bf16_t*)alloc((size_t)768 * 768 * 2);
    bf16_t* wb_cw2 = (bf16_t*)alloc((size_t)256 * 768 * 2);
    bf16_t* wb_cw3 = (bf16_t*)alloc((size_t)64  * 256 * 2);
    bf16_t* wb_tw1 = (bf16_t*)alloc((size_t)384 * 384 * 2);
    bf16_t* wb_tw2 = (bf16_t*)alloc((size_t)256 * 384 * 2);
    bf16_t* wb_tw3 = (bf16_t*)alloc((size_t)64  * 256 * 2);
    bf16_t* wb_cfw = (bf16_t*)alloc((size_t)64  * 64  * 2);
    float*  cfeat   = (float*)alloc((size_t)N * 64 * 4);
    float*  tfeat   = (float*)alloc((size_t)N * 64 * 4);
    float*  cf_lin  = (float*)alloc((size_t)N * 64 * 4);
    float*  mm_full = (float*)alloc((size_t)N * 64 * 4);
    float*  cf_full = (float*)alloc((size_t)N * 64 * 4);
    char*   bufA = alloc((size_t)N * 768 * 2);   // Xc_n -> Xt_n -> cf bf16 (sequential reuse)
    char*   bufB = alloc((size_t)N * 768 * 2);   // H1c -> H1t
    char*   bufC = alloc((size_t)N * 256 * 2);   // H2c -> H2t
    bf16_t* Xc  = (bf16_t*)bufA;
    bf16_t* Xt  = (bf16_t*)bufA;
    bf16_t* cfb = (bf16_t*)bufA;
    bf16_t* H1c = (bf16_t*)bufB;
    bf16_t* H1t = (bf16_t*)bufB;
    bf16_t* H2c = (bf16_t*)bufC;
    bf16_t* H2t = (bf16_t*)bufC;

    auto cvt = [&](const float* s, bf16_t* d, int n) {
        cvt_f32_bf16_kernel<<<(n + 255) / 256, 256, 0, stream>>>(s, d, n);
    };

    const int MG = (N + 255) / 256;  // 79 M-blocks (tile 256)

    // ---- weight conversion ----
    cvt(c_w1, wb_cw1, 768 * 768);
    cvt(c_w2, wb_cw2, 256 * 768);
    cvt(c_w3, wb_cw3, 64 * 256);
    cvt(t_w1, wb_tw1, 384 * 384);
    cvt(t_w2, wb_tw2, 256 * 384);
    cvt(t_w3, wb_tw3, 64 * 256);
    cvt(cf_w, wb_cfw, 64 * 64);

    // ---- content branch: l2 -> 768 -> 768 -> 256 -> 64 ----
    l2norm_rows_kernel<<<N, 256, 0, stream>>>(content, Xc, 768);
    gemm_bf16_wmma_kernel<true,  true ><<<dim3(MG, 12), 256, 0, stream>>>(Xc,  wb_cw1, c_b1, H1c,   N, 768, 768);
    gemm_bf16_wmma_kernel<true,  true ><<<dim3(MG, 4),  256, 0, stream>>>(H1c, wb_cw2, c_b2, H2c,   N, 256, 768);
    gemm_bf16_wmma_kernel<false, false><<<dim3(MG, 1),  256, 0, stream>>>(H2c, wb_cw3, c_b3, cfeat, N, 64,  256);

    // ---- text branch: l2 -> 384 -> 384 -> 256 -> 64 ----
    l2norm_rows_kernel<<<N, 256, 0, stream>>>(text, Xt, 384);
    gemm_bf16_wmma_kernel<true,  true ><<<dim3(MG, 6),  256, 0, stream>>>(Xt,  wb_tw1, t_b1, H1t,   N, 384, 384);
    gemm_bf16_wmma_kernel<true,  true ><<<dim3(MG, 4),  256, 0, stream>>>(H1t, wb_tw2, t_b2, H2t,   N, 256, 384);
    gemm_bf16_wmma_kernel<false, false><<<dim3(MG, 1),  256, 0, stream>>>(H2t, wb_tw3, t_b3, tfeat, N, 64,  256);

    // ---- cf projection ----
    cvt(cf_feat, cfb, N * 64);
    gemm_bf16_wmma_kernel<false, false><<<dim3(MG, 1),  256, 0, stream>>>(cfb, wb_cfw, cf_b, cf_lin, N, 64, 64);

    // ---- combine modalities + cold-item scatter ----
    mm_combine_kernel<<<(N + 7) / 8, 256, 0, stream>>>(cfeat, tfeat, cf_lin, cold, mm_full, cf_full, N);

    // ---- fused gather / L2 / LayerNorm / attention ----
    fused_attn_kernel<<<(TOK + 7) / 8, 256, 0, stream>>>(seq, mm_full, item_emb, cf_full,
                                                         w_q, w_k, w_v, out, TOK);
}
